// GeneralizedLinearAttention_72078141161989
// MI455X (gfx1250) — matrix operand order repaired and checked by reading, not compiled, not run on hardware
//
#include <hip/hip_runtime.h>
#include <hip/hip_bf16.h>

typedef __attribute__((ext_vector_type(16))) __bf16 v16bf;
typedef __attribute__((ext_vector_type(8)))  float  v8f;

constexpr int N_ = 4, L_ = 4096, H_ = 16, D_ = 64;
constexpr int NH = N_ * H_;
constexpr int SCHUNKS = 8;
constexpr int SCHUNK  = L_ / SCHUNKS;            // 512
constexpr int NHSTRIDE = 2 * D_ * D_ + 2 * D_;   // 8320 floats per (n,h): KV1,KV2,Ksum1,Ksum2
constexpr float EPS_ = 1e-6f;

__device__ __forceinline__ float fm_elu1(float x) { return x > 0.f ? x + 1.f : __expf(x); }
__device__ __forceinline__ float fm_relu(float x) { return x > 0.f ? x : 0.f; }

// -------------------------------------------------------------------------
// Stage 1: per (n,h,s-chunk) compute partial KV_fm (64x64) and Ksum_fm (64)
//   KV[m][d] = sum_s fm(K[s,d])*kl[s] * V[s,m]
// WMMA: D(16x16 m,d) = A(16x32: m x s, from V^T) * B(32x16: s x d, from K)
// -------------------------------------------------------------------------
constexpr int S1PAD = 40;   // LDS row stride (elements); 80B keeps 16B alignment

__global__ __launch_bounds__(256)
void kv_partial_kernel(const float* __restrict__ keys,
                       const float* __restrict__ values,
                       const float* __restrict__ klen,
                       float* __restrict__ part)      // [SCHUNKS][NH][NHSTRIDE]
{
    __shared__ __bf16 Kt1[D_][S1PAD];   // Kt[d][s] for fm1 (elu+1)
    __shared__ __bf16 Kt2[D_][S1PAD];   // fm2 (relu)
    __shared__ __bf16 Vt [D_][S1PAD];   // Vt[m][s]

    const int bid   = blockIdx.x;
    const int chunk = bid % SCHUNKS;
    const int nh    = bid / SCHUNKS;
    const int n = nh / H_, h = nh % H_;
    const int t     = threadIdx.x;
    const int wave  = t >> 5;
    const int lane  = t & 31;
    const int fmsel = wave >> 2;          // waves 0-3: fm1, waves 4-7: fm2
    const int mt    = wave & 3;           // m-tile 0..3
    const int lhalf = lane >> 4, l16 = lane & 15;

    v8f acc[4];
#pragma unroll
    for (int dt = 0; dt < 4; ++dt)
#pragma unroll
        for (int r = 0; r < 8; ++r) acc[dt][r] = 0.f;

    float ks = 0.f;                       // Ksum partial (threads < 128)

    const int srow = t >> 3;              // 0..31 : s-row this thread converts
    const int d0   = (t & 7) * 8;         // 8 contiguous d per thread

    const int sbeg = chunk * SCHUNK;
    for (int s0 = sbeg; s0 < sbeg + SCHUNK; s0 += 32) {
        // ---- load 32 rows of K and V, apply feature maps, transpose to LDS bf16
        {
            const int s = s0 + srow;
            const float kl = klen[n * L_ + s];
            const float* kp = keys   + (((long)n * L_ + s) * H_ + h) * D_ + d0;
            const float* vp = values + (((long)n * L_ + s) * H_ + h) * D_ + d0;
#pragma unroll
            for (int j = 0; j < 8; ++j) {
                float kv = kp[j];
                Kt1[d0 + j][srow] = (__bf16)(fm_elu1(kv) * kl);
                Kt2[d0 + j][srow] = (__bf16)(fm_relu(kv) * kl);
                Vt [d0 + j][srow] = (__bf16)vp[j];
            }
        }
        __syncthreads();

        // ---- Ksum accumulation (one d per thread, threads 0..127)
        if (t < 64) {
#pragma unroll
            for (int j = 0; j < 32; ++j) ks += (float)Kt1[t][j];
        } else if (t < 128) {
#pragma unroll
            for (int j = 0; j < 32; ++j) ks += (float)Kt2[t - 64][j];
        }

        // ---- WMMA: A tile (V^T rows m), reused for 4 B tiles (K cols d)
        v16bf a;
        {
            const int m = mt * 16 + l16;
            const int base = lhalf * 8;    // 16-bit A layout: two 8-elem K runs
#pragma unroll
            for (int i = 0; i < 8; ++i) a[i]     = Vt[m][base + i];
#pragma unroll
            for (int i = 0; i < 8; ++i) a[8 + i] = Vt[m][16 + base + i];
        }
#pragma unroll
        for (int dt = 0; dt < 4; ++dt) {
            v16bf b;
            const int d  = dt * 16 + l16;
            const int sb = lhalf * 8;      // 16-bit B layout: two 8-elem K runs
#pragma unroll
            for (int i = 0; i < 16; ++i) {
                const int si = (i < 8) ? (sb + i) : (sb + 8 + i);
                b[i] = fmsel ? Kt2[d][si] : Kt1[d][si];
            }
            acc[dt] = __builtin_amdgcn_wmma_f32_16x16x32_bf16(
                false, a, false, b, (short)0, acc[dt], false, false);
        }
        __syncthreads();
    }

    // ---- write partials (full overwrite of this block's slice)
    float* outp = part + ((long)chunk * NH + nh) * NHSTRIDE;
#pragma unroll
    for (int dt = 0; dt < 4; ++dt)
#pragma unroll
        for (int r = 0; r < 8; ++r) {
            const int m = mt * 16 + r + 8 * lhalf;   // C/D layout: M=r+8*half
            const int d = dt * 16 + l16;             //             N=lane&15
            outp[fmsel * 4096 + m * 64 + d] = acc[dt][r];
        }
    if (t < 128) outp[8192 + t] = ks;
}

// -------------------------------------------------------------------------
// Reduce the 8 s-chunk partials
// -------------------------------------------------------------------------
__global__ __launch_bounds__(256)
void kv_reduce_kernel(const float* __restrict__ part, float* __restrict__ kvws)
{
    const long i = (long)blockIdx.x * 256 + threadIdx.x;  // < NH*NHSTRIDE
    float s = 0.f;
#pragma unroll
    for (int c = 0; c < SCHUNKS; ++c) s += part[(long)c * NH * NHSTRIDE + i];
    kvws[i] = s;
}

// -------------------------------------------------------------------------
// Stage 2: out[l,m] = sum_fm z_fm[l]*pi_fm[l] * (A_fm(Q) @ KV_fm^T)[l,m]
// WMMA: D(16x16 l,m) = A(16x32: l x d, fm(Q)) * B(32x16: d x m, KV^T)
// -------------------------------------------------------------------------
constexpr int S2PAD = 72;   // 144B row stride, 16B aligned

__global__ __launch_bounds__(256)
void out_kernel(const float* __restrict__ queries,
                const float* __restrict__ piproj,   // [H][64][2]
                const float* __restrict__ kvws,     // [NH][NHSTRIDE]
                float* __restrict__ outg)
{
    __shared__ __bf16 KVb1[D_][S2PAD];
    __shared__ __bf16 KVb2[D_][S2PAD];
    __shared__ float  ks1[D_], ks2[D_], pp[2 * D_];

    const int bid    = blockIdx.x;
    const int lchunk = bid & 15;          // 16 chunks of 256 rows
    const int nh     = bid >> 4;
    const int n = nh / H_, h = nh % H_;
    const int t = threadIdx.x, wave = t >> 5, lane = t & 31;
    const int lhalf = lane >> 4, l16 = lane & 15;

    const float* kvbase = kvws + (long)nh * NHSTRIDE;
#pragma unroll
    for (int j = 0; j < 16; ++j) {        // 4096 elems each, 16 per thread
        const int i = t * 16 + j;
        const int m = i >> 6, d = i & 63;
        KVb1[m][d] = (__bf16)kvbase[i];
        KVb2[m][d] = (__bf16)kvbase[4096 + i];
    }
    if (t < 64)  { ks1[t] = kvbase[8192 + t]; ks2[t] = kvbase[8256 + t]; }
    if (t >= 128) pp[t - 128] = piproj[h * 128 + (t - 128)];
    __syncthreads();

#pragma unroll
    for (int sub = 0; sub < 2; ++sub) {   // each wave: two 16-row l-tiles
        const int l0 = lchunk * 256 + wave * 32 + sub * 16;
        const int l  = l0 + l16;
        const float* qp = queries + (((long)n * L_ + l) * H_ + h) * D_;
        const int base = lhalf * 8;

        v16bf A1k0, A1k1, A2k0, A2k1;
        float s1 = 0.f, s2 = 0.f, t0 = 0.f, t1 = 0.f;
#pragma unroll
        for (int i = 0; i < 16; ++i) {
            const int d = (i < 8) ? (base + i) : (16 + base + (i - 8));
            float qv = qp[d];
            float f1 = fm_elu1(qv), f2 = fm_relu(qv);
            A1k0[i] = (__bf16)f1;  A2k0[i] = (__bf16)f2;
            s1 += f1 * ks1[d];     s2 += f2 * ks2[d];
            t0 += qv * pp[2 * d];  t1 += qv * pp[2 * d + 1];
            const int d2 = d + 32;
            float qw = qp[d2];
            float g1 = fm_elu1(qw), g2 = fm_relu(qw);
            A1k1[i] = (__bf16)g1;  A2k1[i] = (__bf16)g2;
            s1 += g1 * ks1[d2];    s2 += g2 * ks2[d2];
            t0 += qw * pp[2 * d2]; t1 += qw * pp[2 * d2 + 1];
        }
        // combine the two d-halves of the wave; lanes r and r+16 both get row r
        s1 += __shfl_xor(s1, 16, 32);
        s2 += __shfl_xor(s2, 16, 32);
        t0 += __shfl_xor(t0, 16, 32);
        t1 += __shfl_xor(t1, 16, 32);
        const float z1 = 1.f / (s1 + EPS_), z2 = 1.f / (s2 + EPS_);
        const float mx = fmaxf(t0, t1);
        const float e0 = __expf(t0 - mx), e1 = __expf(t1 - mx);
        const float inv = 1.f / (e0 + e1);
        const float c1 = z1 * e0 * inv;   // z * pi per row (held in lanes 0..15)
        const float c2 = z2 * e1 * inv;

#pragma unroll
        for (int mtile = 0; mtile < 4; ++mtile) {
            v16bf b10, b11, b20, b21;
            const int m  = mtile * 16 + l16;
            const int db = lhalf * 8;
#pragma unroll
            for (int i = 0; i < 16; ++i) {
                const int di = (i < 8) ? (db + i) : (db + 8 + i);
                b10[i] = KVb1[m][di];        b20[i] = KVb2[m][di];
                b11[i] = KVb1[m][32 + di];   b21[i] = KVb2[m][32 + di];
            }
            v8f acc1, acc2;
#pragma unroll
            for (int r = 0; r < 8; ++r) { acc1[r] = 0.f; acc2[r] = 0.f; }
            acc1 = __builtin_amdgcn_wmma_f32_16x16x32_bf16(false, A1k0, false, b10, (short)0, acc1, false, false);
            acc1 = __builtin_amdgcn_wmma_f32_16x16x32_bf16(false, A1k1, false, b11, (short)0, acc1, false, false);
            acc2 = __builtin_amdgcn_wmma_f32_16x16x32_bf16(false, A2k0, false, b20, (short)0, acc2, false, false);
            acc2 = __builtin_amdgcn_wmma_f32_16x16x32_bf16(false, A2k1, false, b21, (short)0, acc2, false, false);
#pragma unroll
            for (int r = 0; r < 8; ++r) {
                const int row = r + 8 * lhalf;
                const float f = acc1[r] * __shfl(c1, row, 32)
                              + acc2[r] * __shfl(c2, row, 32);
                outg[(((long)n * L_ + (l0 + row)) * H_ + h) * D_ + mtile * 16 + l16] = f;
            }
        }
    }
}

// -------------------------------------------------------------------------
extern "C" void kernel_launch(void* const* d_in, const int* in_sizes, int n_in,
                              void* d_out, int out_size, void* d_ws, size_t ws_size,
                              hipStream_t stream)
{
    (void)in_sizes; (void)n_in; (void)out_size; (void)ws_size;
    const float* queries = (const float*)d_in[0];
    const float* keys    = (const float*)d_in[1];
    const float* values  = (const float*)d_in[2];
    const float* klen    = (const float*)d_in[3];
    const float* piproj  = (const float*)d_in[4];
    float* out = (float*)d_out;

    float* part = (float*)d_ws;                               // [8][64][8320]
    float* kvws = part + (size_t)SCHUNKS * NH * NHSTRIDE;     // [64][8320]

    kv_partial_kernel<<<NH * SCHUNKS, 256, 0, stream>>>(keys, values, klen, part);
    kv_reduce_kernel<<<(NH * NHSTRIDE) / 256, 256, 0, stream>>>(part, kvws);
    out_kernel<<<NH * 16, 256, 0, stream>>>(queries, piproj, kvws, out);
}
